// new_17669495455991
// MI455X (gfx1250) — matrix operand order repaired and checked by reading, not compiled, not run on hardware
//
#include <hip/hip_runtime.h>
#include <hip/hip_bf16.h>
#include <cstdint>
#include <cstddef>

// ---------------------------------------------------------------------------
// MI455X / gfx1250: bf16 WMMA fused MLP + invariant pooling (v2)
//  - fused gather + GEMM1(324->256)+ReLU + GEMM2(256->128) via v_wmma bf16
//  - inputs read with NT hint so the 128MB h2 intermediate stays in 192MB L2
//  - single pooling pass derives row/col sums as byproducts (no extra passes)
// ---------------------------------------------------------------------------

typedef __bf16 bf16_t;
typedef bf16_t v16bf __attribute__((ext_vector_type(16)));
typedef bf16_t v8bf  __attribute__((ext_vector_type(8)));
typedef float  v8f   __attribute__((ext_vector_type(8)));

#define NTT  27
#define NC   128
#define NB   16
#define K1   324          // 27 * 12
#define K1P  352          // padded to 11 * 32
#define N1   256
#define N2   128
#define LDA  360          // LDS row stride (bf16) for A tile  (720B, 16B-mult)
#define LDH  264          // LDS row stride (bf16) for layer1   (528B, 16B-mult)

__device__ __forceinline__ bf16_t f2bf(float f) {
    union { float f; uint32_t u; } v; v.f = f;
    uint32_t r = (v.u + 0x7FFFu + ((v.u >> 16) & 1u)) >> 16;   // RNE truncate
    union { uint16_t s; bf16_t b; } o; o.s = (uint16_t)r;
    return o.b;
}

union AFrag { v16bf v; v8bf h[2]; };

// --------------------------------------------------------------------------
// Pre-transpose + bf16-convert weights into B-fragment-friendly [N][K] rows.
// --------------------------------------------------------------------------
__global__ void prep_weights_kernel(const float* __restrict__ W1,
                                    const float* __restrict__ W2,
                                    bf16_t* __restrict__ W1bf,
                                    bf16_t* __restrict__ W2bf) {
    int i = blockIdx.x * blockDim.x + threadIdx.x;
    const int tot1 = N1 * K1P;            // 90112
    if (i < tot1) {
        int n = i / K1P, k = i % K1P;
        W1bf[i] = (k < K1) ? f2bf(W1[k * N1 + n]) : f2bf(0.0f);
    }
    int j = i - tot1;
    if (j >= 0 && j < N2 * 256) {
        int n = j / 256, k = j % 256;
        W2bf[j] = f2bf(W2[k * N2 + n]);
    }
}

__global__ void zero_kernel(float* __restrict__ p, int n) {
    int i = blockIdx.x * blockDim.x + threadIdx.x;
    if (i < n) p[i] = 0.0f;
}

// --------------------------------------------------------------------------
// Fused gather + GEMM1 + ReLU + GEMM2 (bf16 WMMA).
// One block = sample b, row a, 16 consecutive cols. 16384 blocks x 128 thr.
// --------------------------------------------------------------------------
__global__ void __launch_bounds__(128)
fused_gemm_kernel(const float* __restrict__ f0, const float* __restrict__ f1,
                  const float* __restrict__ f2, const float* __restrict__ f3,
                  const bf16_t* __restrict__ W1bf, const float* __restrict__ b1,
                  const bf16_t* __restrict__ W2bf, const float* __restrict__ b2,
                  float* __restrict__ h2out) {
    __shared__ __align__(16) bf16_t Abuf[16 * LDA];
    __shared__ __align__(16) bf16_t H1[16 * LDH];

    const int blk = blockIdx.x;
    const int b   = blk >> 10;
    const int a   = (blk >> 3) & 127;
    const int c0  = (blk & 7) << 4;

    const int tid  = threadIdx.x;
    const int lane = tid & 31;
    const int wave = tid >> 5;
    const int m    = lane & 15;            // M (for A) or N-within-tile (for B/C)
    const int hi   = (lane >> 4) & 1;      // lane half selects K/M sub-block

    // zero the K-pad region [324, LDA)
    for (int i = tid; i < 16 * (LDA - K1); i += 128) {
        int r = i / (LDA - K1), k = K1 + i % (LDA - K1);
        Abuf[r * LDA + k] = f2bf(0.0f);
    }
    // gather A tile: fv[b, a, c, t*12 + s*3 + e]; NT loads (inputs read once;
    // keep h2 resident in L2)
    for (int i = tid; i < 16 * NTT * 4; i += 128) {
        int row = i & 15;
        int q   = i >> 4;                  // 0..107
        int t   = q >> 2;
        int s   = q & 3;
        const float* src = (s == 0) ? f0 : (s == 1) ? f1 : (s == 2) ? f2 : f3;
        const float* p = src + ((((size_t)b * NTT + t) * NC + a) * NC + (c0 + row)) * 3;
        float e0 = __builtin_nontemporal_load(p + 0);
        float e1 = __builtin_nontemporal_load(p + 1);
        float e2 = __builtin_nontemporal_load(p + 2);
        int k = t * 12 + s * 3;
        Abuf[row * LDA + k + 0] = f2bf(e0);
        Abuf[row * LDA + k + 1] = f2bf(e1);
        Abuf[row * LDA + k + 2] = f2bf(e2);
    }
    __syncthreads();

    // ---- GEMM1: [16 x 352] x [352 x 256], wave handles 4 N-tiles ----
    v8f acc[4];
    #pragma unroll
    for (int j = 0; j < 4; ++j)
        acc[j] = v8f{0.f, 0.f, 0.f, 0.f, 0.f, 0.f, 0.f, 0.f};

    for (int kc = 0; kc < K1P; kc += 32) {
        AFrag af;
        af.h[0] = *(const v8bf*)&Abuf[m * LDA + kc + hi * 16];
        af.h[1] = *(const v8bf*)&Abuf[m * LDA + kc + hi * 16 + 8];
        #pragma unroll
        for (int j = 0; j < 4; ++j) {
            int n = (wave * 4 + j) * 16 + m;
            v16bf bfrag = *(const v16bf*)&W1bf[(size_t)n * K1P + kc + hi * 16];
            acc[j] = __builtin_amdgcn_wmma_f32_16x16x32_bf16(
                false, af.v, false, bfrag, (short)0, acc[j], false, false);
        }
    }

    // bias + ReLU, re-lay-out C->A through LDS (bf16)
    #pragma unroll
    for (int j = 0; j < 4; ++j) {
        int n = (wave * 4 + j) * 16 + m;
        float bias = b1[n];
        #pragma unroll
        for (int r = 0; r < 8; ++r) {
            float v = acc[j][r] + bias;
            v = v > 0.f ? v : 0.f;
            H1[(r + hi * 8) * LDH + n] = f2bf(v);
        }
    }
    __syncthreads();

    // ---- GEMM2: [16 x 256] x [256 x 128], wave handles 2 N-tiles ----
    v8f acc2[2];
    #pragma unroll
    for (int j = 0; j < 2; ++j)
        acc2[j] = v8f{0.f, 0.f, 0.f, 0.f, 0.f, 0.f, 0.f, 0.f};

    for (int kc = 0; kc < 256; kc += 32) {
        AFrag af;
        af.h[0] = *(const v8bf*)&H1[m * LDH + kc + hi * 16];
        af.h[1] = *(const v8bf*)&H1[m * LDH + kc + hi * 16 + 8];
        #pragma unroll
        for (int j = 0; j < 2; ++j) {
            int n = (wave * 2 + j) * 16 + m;
            v16bf bfrag = *(const v16bf*)&W2bf[(size_t)n * 256 + kc + hi * 16];
            acc2[j] = __builtin_amdgcn_wmma_f32_16x16x32_bf16(
                false, af.v, false, bfrag, (short)0, acc2[j], false, false);
        }
    }

    // write layer2 (+ bias, no relu) as f32: h2[b][a][c0+M][z] (RT -> L2)
    size_t base = (((size_t)b * NC + a) * NC + c0) * N2;
    #pragma unroll
    for (int j = 0; j < 2; ++j) {
        int n = (wave * 2 + j) * 16 + m;
        float bias = b2[n];
        #pragma unroll
        for (int r = 0; r < 8; ++r) {
            int M = r + hi * 8;
            h2out[base + (size_t)M * N2 + n] = acc2[j][r] + bias;
        }
    }
}

// --------------------------------------------------------------------------
// Single-pass pooling: block (b,a) walks row a and column a of x[.,.,z];
// row/col sums fall out of the same loop. Pad channel is analytic constants.
// --------------------------------------------------------------------------
__global__ void pool_kernel(const float* __restrict__ h2, float* __restrict__ h8) {
    int blk = blockIdx.x;                  // b*128 + a
    int b = blk >> 7, a = blk & 127;
    int z = threadIdx.x;
    const float* base  = h2 + ((size_t)b * NC + a) * NC * N2 + z;     // x[a,c,z]
    const float* baseT = h2 + ((size_t)b * NC * NC + a) * N2 + z;     // x[c,a,z]
    float s_row = 0.f, s_col = 0.f, s_sq = 0.f, s_xt = 0.f, diag = 0.f;
    for (int c = 0; c < NC; ++c) {
        float x  = base[(size_t)c * N2];
        float xt = baseT[(size_t)c * NC * N2];
        s_row += x; s_col += xt;
        s_sq  += x * x; s_xt += x * xt;
        if (c == a) diag = x;
    }
    float* o = h8 + (size_t)b * 8 * N2 + z;
    atomicAdd(o + 0 * N2, diag);            // sum diag
    atomicAdd(o + 1 * N2, s_row);           // sum all
    atomicAdd(o + 2 * N2, diag * diag);     // sum diag^2
    atomicAdd(o + 3 * N2, diag * s_row);    // sum diag*row
    atomicAdd(o + 4 * N2, s_sq);            // sum x^2
    atomicAdd(o + 5 * N2, s_xt);            // sum x*x^T
    atomicAdd(o + 6 * N2, s_row * s_row);   // sum row^2
    atomicAdd(o + 7 * N2, s_row * s_col);   // sum row*col
}

// --------------------------------------------------------------------------
// Head: const-denominator normalize, vector_log, GEMM3+ReLU, GEMM4, tanh*8.
// --------------------------------------------------------------------------
__global__ void __launch_bounds__(256)
head_kernel(const float* __restrict__ h8,
            const float* __restrict__ W3, const float* __restrict__ b3,
            const float* __restrict__ W4, const float* __restrict__ b4,
            float* __restrict__ out) {
    __shared__ float u[1024];
    __shared__ float l3[256];
    const int tid = threadIdx.x;
    // denominators from the analytic all-ones pad channel
    const float denom[8] = {128.f, 16384.f, 128.f, 16384.f,
                            16384.f, 16384.f, 2097152.f, 2097152.f};
    for (int b = 0; b < NB; ++b) {
        for (int i = tid; i < 1024; i += 256) {
            float x = h8[b * 1024 + i] / (denom[i >> 7] + 1e-20f);
            float s = (x < 0.f) ? -1.f : 1.f;
            u[i] = s * log1pf(fabsf(x)) * 0.1f;
        }
        __syncthreads();
        float acc = b3[tid];
        for (int k = 0; k < 1024; ++k) acc += u[k] * W3[k * 256 + tid];
        l3[tid] = acc > 0.f ? acc : 0.f;
        __syncthreads();
        if (tid < 2) {
            float a2 = b4[tid];
            for (int k = 0; k < 256; ++k) a2 += l3[k] * W4[k * 2 + tid];
            out[b * 2 + tid] = tanhf(a2) * 8.f;
        }
        __syncthreads();
    }
}

// --------------------------------------------------------------------------
extern "C" void kernel_launch(void* const* d_in, const int* in_sizes, int n_in,
                              void* d_out, int out_size, void* d_ws, size_t ws_size,
                              hipStream_t stream) {
    const float* f0 = (const float*)d_in[0];
    const float* f1 = (const float*)d_in[1];
    const float* f2 = (const float*)d_in[2];
    const float* f3 = (const float*)d_in[3];
    const float* W1 = (const float*)d_in[4];
    const float* b1 = (const float*)d_in[5];
    const float* W2 = (const float*)d_in[6];
    const float* b2 = (const float*)d_in[7];
    const float* W3 = (const float*)d_in[8];
    const float* b3 = (const float*)d_in[9];
    const float* W4 = (const float*)d_in[10];
    const float* b4 = (const float*)d_in[11];
    float* out = (float*)d_out;

    // workspace layout (all offsets 256B-aligned)
    char* ws = (char*)d_ws;
    float*  h2   = (float*)ws;                                  // 134,217,728 B
    size_t  off  = (size_t)NB * NC * NC * N2 * sizeof(float);
    bf16_t* W1bf = (bf16_t*)(ws + off);  off += (size_t)N1 * K1P * sizeof(bf16_t); // 180,224
    bf16_t* W2bf = (bf16_t*)(ws + off);  off += (size_t)N2 * 256 * sizeof(bf16_t); //  65,536
    float*  h8   = (float*)(ws + off);   off += (size_t)NB * 8 * N2 * sizeof(float);
    (void)ws_size; (void)in_sizes; (void)n_in; (void)out_size;

    // 1) weights -> bf16 transposed layout (L2-resident thereafter)
    prep_weights_kernel<<<dim3((N1 * K1P + N2 * 256 + 255) / 256), dim3(256), 0, stream>>>(
        W1, W2, W1bf, W2bf);
    // 2) zero pooling accumulators
    zero_kernel<<<dim3((NB * 8 * N2 + 255) / 256), dim3(256), 0, stream>>>(h8, NB * 8 * N2);
    // 3) fused gather + GEMM1 + ReLU + GEMM2 (bf16 WMMA)
    fused_gemm_kernel<<<dim3(NB * NC * (NC / 16)), dim3(128), 0, stream>>>(
        f0, f1, f2, f3, W1bf, b1, W2bf, b2, h2);
    // 4) single-pass invariants (row/col sums derived in-loop)
    pool_kernel<<<dim3(NB * NC), dim3(N2), 0, stream>>>(h2, h8);
    // 5) head
    head_kernel<<<dim3(1), dim3(256), 0, stream>>>(h8, W3, b3, W4, b4, out);
}
